// NaInsBlock_66108136620626
// MI455X (gfx1250) — matrix operand order repaired and checked by reading, not compiled, not run on hardware
//
#include <hip/hip_runtime.h>
#include <hip/hip_bf16.h>

typedef __attribute__((ext_vector_type(16))) _Float16 v16h;
typedef __attribute__((ext_vector_type(8)))  _Float16 v8h;
typedef __attribute__((ext_vector_type(8)))  float    v8f;

#define NVOX 32768   // 32*32*32
#define CCH  128
#define NQKV 384

// ---------------- weight f32 -> f16 convert ----------------
__global__ void cvt_f16_kernel(const float* __restrict__ src, _Float16* __restrict__ dst, int n) {
    int i = blockIdx.x * blockDim.x + threadIdx.x;
    if (i < n) dst[i] = (_Float16)src[i];
}

// ---------------- LayerNorm over channels ----------------
// x is channel-major (C, 32768). One thread per voxel; channel loop is coalesced
// across lanes. Writes h voxel-major (32768 x 128) f16, 16B stores per lane.
__global__ void ln_kernel(const float* __restrict__ x, const float* __restrict__ gamma,
                          const float* __restrict__ beta, _Float16* __restrict__ h) {
    int v = blockIdx.x * blockDim.x + threadIdx.x;
    float s = 0.f, ss = 0.f;
    for (int c = 0; c < CCH; ++c) {
        float val = x[c * NVOX + v];
        s  += val;
        ss += val * val;
    }
    float mu  = s * (1.0f / CCH);
    float var = ss * (1.0f / CCH) - mu * mu;
    float r   = rsqrtf(var + 1e-5f);
    for (int c0 = 0; c0 < CCH; c0 += 8) {
        v8h o;
        #pragma unroll
        for (int j = 0; j < 8; ++j) {
            int c = c0 + j;
            o[j] = (_Float16)((x[c * NVOX + v] - mu) * r * gamma[c] + beta[c]);
        }
        *(v8h*)(h + v * CCH + c0) = o;
    }
}

// ---------------- QKV GEMM: (32768x128)f16 @ (384x128)^T f16 -> (32768x384)f32
// One wave per 16(M) x 64(N) strip: 4 accumulators, A fragment loaded once per
// K-step and reused across 4 WMMAs. K=128 in 4 f16 16x16x32 steps.
__global__ void qkv_gemm_kernel(const _Float16* __restrict__ A, const _Float16* __restrict__ W,
                                const float* __restrict__ bias, float* __restrict__ out) {
    int warp = threadIdx.x >> 5;
    int lane = threadIdx.x & 31;
    int mgrp = blockIdx.x & 255;               // 256 M groups of 8 tiles
    int ng   = blockIdx.x >> 8;                // 6 N groups of 4 tiles
    int mt   = mgrp * 8 + warp;
    int m0 = mt << 4, n0 = ng << 6;
    int g  = lane >> 4;                        // lane half
    int ml = lane & 15;                        // row (A) / col (B) within tile

    const _Float16* arow = A + (size_t)(m0 + ml) * CCH;  // A: M row, K contiguous
    const _Float16* wrow0 = W + (size_t)(n0 + ml) * CCH; // 4 weight tiles, 16 rows apart

    // prefetch next M strip of A (streams h through WGP$)
    __builtin_prefetch(arow + 16 * CCH, 0, 1);

    v8f acc0 = {}, acc1 = {}, acc2 = {}, acc3 = {};
    #pragma unroll
    for (int ks = 0; ks < 4; ++ks) {
        int kb = ks * 32;
        // A lane (m + 16g): VGPR0-3 <- K = kb+8g .. +7 ; VGPR4-7 <- K = kb+16+8g .. +7
        v8h a_lo = *(const v8h*)(arow + kb + 8 * g);
        v8h a_hi = *(const v8h*)(arow + kb + 16 + 8 * g);
        v16h av;
        #pragma unroll
        for (int j = 0; j < 8; ++j) { av[j] = a_lo[j]; av[j + 8] = a_hi[j]; }
        #pragma unroll
        for (int t = 0; t < 4; ++t) {
            // B lane (n + 16g): VGPR0-3 <- K = kb+8g .. +7 ; VGPR4-7 <- K = kb+16+8g .. +7
            const _Float16* wrow = wrow0 + (size_t)(t * 16) * CCH;
            v8h b_lo = *(const v8h*)(wrow + kb + 8 * g);
            v8h b_hi = *(const v8h*)(wrow + kb + 16 + 8 * g);
            v16h bv;
            #pragma unroll
            for (int j = 0; j < 8; ++j) { bv[j] = b_lo[j]; bv[j + 8] = b_hi[j]; }
            v8f& acc = (t == 0) ? acc0 : (t == 1) ? acc1 : (t == 2) ? acc2 : acc3;
            acc = __builtin_amdgcn_wmma_f32_16x16x32_f16(false, av, false, bv,
                                                         (short)0, acc, false, false);
        }
    }
    // D: VGPR r -> M = m0 + r + 8g, N = n0 + 16t + ml
    #pragma unroll
    for (int t = 0; t < 4; ++t) {
        v8f& acc = (t == 0) ? acc0 : (t == 1) ? acc1 : (t == 2) ? acc2 : acc3;
        int n = n0 + t * 16 + ml;
        float bn = bias[n];
        #pragma unroll
        for (int r = 0; r < 8; ++r) {
            int m = m0 + r + 8 * g;
            out[(size_t)m * NQKV + n] = acc[r] + bn;
        }
    }
}

// ---------------- 3x3x3 neighborhood attention ----------------
// One wave per (voxel, head); lane = head channel (hd == warpSize == 32).
__global__ void attn_kernel(const float* __restrict__ qkv, _Float16* __restrict__ attn_out) {
    int lane  = threadIdx.x & 31;
    int wv    = (blockIdx.x * blockDim.x + threadIdx.x) >> 5;
    int voxel = wv >> 2;
    int head  = wv & 3;
    int xi = voxel >> 10, yi = (voxel >> 5) & 31, zi = voxel & 31;
    int sx = min(max(xi - 1, 0), 29);
    int sy = min(max(yi - 1, 0), 29);
    int sz = min(max(zi - 1, 0), 29);

    const float scale = 0.17677669529663687f; // 32^-0.5
    int hc = head * 32 + lane;
    float qf = qkv[(size_t)voxel * NQKV + hc] * scale;

    float logit[27];
    #pragma unroll
    for (int n = 0; n < 27; ++n) {
        int dx = n / 9, dy = (n / 3) % 3, dz = n % 3;
        int nv = ((sx + dx) << 10) + ((sy + dy) << 5) + (sz + dz);
        float p = qf * qkv[(size_t)nv * NQKV + 128 + hc];
        #pragma unroll
        for (int off = 16; off > 0; off >>= 1) p += __shfl_xor(p, off, 32);
        logit[n] = p;   // all lanes hold the full dot product
    }
    float mx = logit[0];
    #pragma unroll
    for (int n = 1; n < 27; ++n) mx = fmaxf(mx, logit[n]);
    float den = 0.f;
    #pragma unroll
    for (int n = 0; n < 27; ++n) { logit[n] = __expf(logit[n] - mx); den += logit[n]; }
    float inv = 1.0f / den;
    float acc = 0.f;
    #pragma unroll
    for (int n = 0; n < 27; ++n) {
        int dx = n / 9, dy = (n / 3) % 3, dz = n % 3;
        int nv = ((sx + dx) << 10) + ((sy + dy) << 5) + (sz + dz);
        acc += logit[n] * inv * qkv[(size_t)nv * NQKV + 256 + hc];
    }
    attn_out[(size_t)voxel * CCH + hc] = (_Float16)acc;
}

// ---------------- Proj GEMM + bias + residual, store (C, X, Y, Z) ----------------
// One wave per 16(M) x 64(N) strip, same register blocking as the QKV GEMM.
__global__ void proj_gemm_kernel(const _Float16* __restrict__ A, const _Float16* __restrict__ W,
                                 const float* __restrict__ bias, const float* __restrict__ xres,
                                 float* __restrict__ out) {
    int warp = threadIdx.x >> 5;
    int lane = threadIdx.x & 31;
    int mgrp = blockIdx.x & 255;               // 256 M groups of 8 tiles
    int ng   = blockIdx.x >> 8;                // 2 N groups of 4 tiles
    int mt   = mgrp * 8 + warp;
    int m0 = mt << 4, n0 = ng << 6;
    int g  = lane >> 4;
    int ml = lane & 15;

    const _Float16* arow  = A + (size_t)(m0 + ml) * CCH;
    const _Float16* wrow0 = W + (size_t)(n0 + ml) * CCH;

    __builtin_prefetch(arow + 16 * CCH, 0, 1);

    v8f acc0 = {}, acc1 = {}, acc2 = {}, acc3 = {};
    #pragma unroll
    for (int ks = 0; ks < 4; ++ks) {
        int kb = ks * 32;
        v8h a_lo = *(const v8h*)(arow + kb + 8 * g);
        v8h a_hi = *(const v8h*)(arow + kb + 16 + 8 * g);
        v16h av;
        #pragma unroll
        for (int j = 0; j < 8; ++j) { av[j] = a_lo[j]; av[j + 8] = a_hi[j]; }
        #pragma unroll
        for (int t = 0; t < 4; ++t) {
            const _Float16* wrow = wrow0 + (size_t)(t * 16) * CCH;
            v8h b_lo = *(const v8h*)(wrow + kb + 8 * g);
            v8h b_hi = *(const v8h*)(wrow + kb + 16 + 8 * g);
            v16h bv;
            #pragma unroll
            for (int j = 0; j < 8; ++j) { bv[j] = b_lo[j]; bv[j + 8] = b_hi[j]; }
            v8f& acc = (t == 0) ? acc0 : (t == 1) ? acc1 : (t == 2) ? acc2 : acc3;
            acc = __builtin_amdgcn_wmma_f32_16x16x32_f16(false, av, false, bv,
                                                         (short)0, acc, false, false);
        }
    }
    #pragma unroll
    for (int t = 0; t < 4; ++t) {
        v8f& acc = (t == 0) ? acc0 : (t == 1) ? acc1 : (t == 2) ? acc2 : acc3;
        int n = n0 + t * 16 + ml;
        float bn = bias[n];
        #pragma unroll
        for (int r = 0; r < 8; ++r) {
            int m = m0 + r + 8 * g;
            size_t idx = (size_t)n * NVOX + m;     // output is channel-major
            out[idx] = acc[r] + bn + xres[idx];
        }
    }
}

extern "C" void kernel_launch(void* const* d_in, const int* in_sizes, int n_in,
                              void* d_out, int out_size, void* d_ws, size_t ws_size,
                              hipStream_t stream) {
    const float* x       = (const float*)d_in[0];
    const float* ln_g    = (const float*)d_in[1];
    const float* ln_b    = (const float*)d_in[2];
    const float* qkv_w   = (const float*)d_in[3];
    const float* qkv_b   = (const float*)d_in[4];
    const float* proj_w  = (const float*)d_in[5];
    const float* proj_b  = (const float*)d_in[6];
    float* out = (float*)d_out;

    char* ws = (char*)d_ws;
    _Float16* h    = (_Float16*)(ws);                                   //  8 MiB
    float*    qkv  = (float*)   (ws + (size_t)8  * 1024 * 1024);        // 48 MiB
    _Float16* attn = (_Float16*)(ws + (size_t)56 * 1024 * 1024);        //  8 MiB
    _Float16* wq   = (_Float16*)(ws + (size_t)64 * 1024 * 1024);        // 96 KiB
    _Float16* wp   = (_Float16*)(ws + (size_t)64 * 1024 * 1024 + 128 * 1024);

    // 1) weight conversion
    cvt_f16_kernel<<<(NQKV * CCH + 255) / 256, 256, 0, stream>>>(qkv_w, wq, NQKV * CCH);
    cvt_f16_kernel<<<(CCH * CCH + 255) / 256, 256, 0, stream>>>(proj_w, wp, CCH * CCH);

    // 2) LayerNorm -> h (f16, voxel-major)
    ln_kernel<<<NVOX / 256, 256, 0, stream>>>(x, ln_g, ln_b, h);

    // 3) QKV GEMM: 256 M-groups x 6 N-groups, 8 waves/block, 16x64 strip/wave
    qkv_gemm_kernel<<<256 * 6, 256, 0, stream>>>(h, wq, qkv_b, qkv);

    // 4) neighborhood attention: 32768 voxels * 4 heads waves
    attn_kernel<<<(NVOX * 4 * 32) / 256, 256, 0, stream>>>(qkv, attn);

    // 5) Proj GEMM + bias + residual: 256 M-groups x 2 N-groups
    proj_gemm_kernel<<<256 * 2, 256, 0, stream>>>(attn, wp, proj_b, x, out);
}
